// TemporalAttentionLayer_4535485464818
// MI455X (gfx1250) — compile-verified, matrix operand order checked by reading, not run on hardware
//
#include <hip/hip_runtime.h>

// ---------------- types ----------------
typedef __bf16 bf16;
typedef __attribute__((ext_vector_type(16))) __bf16 v16bf;
typedef __attribute__((ext_vector_type(8)))  __bf16 v8bf;
typedef __attribute__((ext_vector_type(8)))  float  v8f;

// ---------------- LDS layout (bytes) ----------------
#define TS    264                      // tin/q/k/out/y1 row stride (bf16 elems), 528B rows
#define VTS   72                       // vT row stride (bf16), 144B rows
#define AS    72                       // attn row stride (bf16), 144B rows
#define Y2S   260                      // y2 f32 row stride, 1040B rows

#define OFF_SEG   0
#define OFF_TIN   256
#define OFF_Q     (OFF_TIN + 64*TS*2)          // 34048
#define OFF_K     (OFF_Q   + 64*TS*2)          // 67840
#define OFF_VT    (OFF_K   + 64*TS*2)          // 101632
#define OFF_ATTN  (OFF_VT  + 8*32*VTS*2)       // 138496
#define SMEM_BYTES (OFF_ATTN + 8*64*AS*2)      // 212224
// overlays (phase-disjoint lifetimes)
#define OFF_OUT   OFF_TIN                      // attn output (bf16) reuses tin
#define OFF_Y1    OFF_Q                        // FF1 output (bf16) reuses q
#define OFF_Y2    OFF_K                        // FF2 output (f32) reuses k+vT

// ---------------- scalar conversions ----------------
__device__ __forceinline__ bf16 f2bf(float f) {
  union { float f; unsigned u; } a; a.f = f;
  unsigned r = a.u + 0x7FFFu + ((a.u >> 16) & 1u);       // round-to-nearest-even
  union { unsigned short s; bf16 b; } o; o.s = (unsigned short)(r >> 16);
  return o.b;
}
__device__ __forceinline__ float bf2f(bf16 b) {
  union { unsigned short s; bf16 b; } i; i.b = b;
  union { unsigned u; float f; } o; o.u = ((unsigned)i.s) << 16;
  return o.f;
}

// ---------------- WMMA helpers ----------------
// A-fragment (16x32 bf16, row-major source [row][k], stride in elems):
// lane L holds row (L&15); K chunks: base=(L>>4)*8, elems [k0+base..+7] and [k0+16+base..+7]
__device__ __forceinline__ v16bf ld_frag(const bf16* p, int stride, int row0, int k0) {
  int lane = threadIdx.x & 31;
  const bf16* q = p + (row0 + (lane & 15)) * stride + k0 + ((lane >> 4) << 3);
  v8bf lo = *(const v8bf*)q;
  v8bf hi = *(const v8bf*)(q + 16);
  v16bf r;
#pragma unroll
  for (int i = 0; i < 8; ++i) { r[i] = lo[i]; r[i + 8] = hi[i]; }
  return r;
}

__device__ __forceinline__ v8f wmma_bf16(v16bf a, v16bf b, v8f c) {
  return __builtin_amdgcn_wmma_f32_16x16x32_bf16(false, a, false, b, (short)0, c, false, false);
}

// C-fragment (16x16 f32) -> bf16 LDS, row-major [row][col]
__device__ __forceinline__ void st_c_bf16(bf16* p, int stride, int row0, int col0, v8f c) {
  int lane = threadIdx.x & 31;
  int col = col0 + (lane & 15);
  int r0  = row0 + ((lane >> 4) << 3);
#pragma unroll
  for (int i = 0; i < 8; ++i) p[(r0 + i) * stride + col] = f2bf(c[i]);
}

// ---------------- weight conversion: fp32 -> bf16 (Wq/Wk/Wv transposed, Wc direct) --------
__global__ void wconv_kernel(const float* __restrict__ Wq, const float* __restrict__ Wk,
                             const float* __restrict__ Wv, const float* __restrict__ Wc,
                             bf16* __restrict__ ws) {
  int idx = blockIdx.x * blockDim.x + threadIdx.x;   // 0 .. 4*65536-1
  int mat = idx >> 16;
  int e   = idx & 65535;
  int r = e >> 8, c = e & 255;                       // ws[mat][r][c]
  float v;
  if      (mat == 0) v = Wq[c * 256 + r];            // WqT[d][c] = Wq[c][d]
  else if (mat == 1) v = Wk[c * 256 + r];
  else if (mat == 2) v = Wv[c * 256 + r];
  else               v = Wc[e];                      // Wc already [o][c]
  ws[idx] = f2bf(v);
}

// ---------------- fused attention layer: one block per sample n ----------------
__global__ __launch_bounds__(256, 1)
void attn_fused_kernel(const float* __restrict__ inp, const long long* __restrict__ seg,
                       const float* __restrict__ pos, const bf16* __restrict__ wbf,
                       const float* __restrict__ bc,  const float* __restrict__ g,
                       const float* __restrict__ bln, float* __restrict__ out) {
  extern __shared__ char smem[];
  int*  s_seg  = (int*) (smem + OFF_SEG);
  bf16* s_tin  = (bf16*)(smem + OFF_TIN);
  bf16* s_q    = (bf16*)(smem + OFF_Q);
  bf16* s_k    = (bf16*)(smem + OFF_K);
  bf16* s_vt   = (bf16*)(smem + OFF_VT);
  bf16* s_attn = (bf16*)(smem + OFF_ATTN);
  bf16* s_out  = (bf16*)(smem + OFF_OUT);
  bf16* s_y1   = (bf16*)(smem + OFF_Y1);
  float* s_y2  = (float*)(smem + OFF_Y2);

  const int tid  = threadIdx.x;
  const int wv   = tid >> 5;
  const int lane = tid & 31;
  const int n    = blockIdx.x;

  const bf16* wq = wbf;                 // [d][c] bf16
  const bf16* wk = wbf + 65536;
  const bf16* wvv = wbf + 131072;
  const bf16* wc = wbf + 196608;        // [o][c] bf16

  // ---- phase 0/1: seg + tin = inputs + pos_emb (fp32 read once, bf16 to LDS) ----
  if (tid < 64) s_seg[tid] = (int)seg[tid];
  const float* ibase = inp + (size_t)n * 16384;
#pragma unroll
  for (int e = tid; e < 4096; e += 256) {           // float4 granules
    int t = e >> 6, c4 = (e & 63) << 2;
    float4 a  = *(const float4*)(ibase + t * 256 + c4);
    float4 pz = *(const float4*)(pos   + t * 256 + c4);
    bf16* d = s_tin + t * TS + c4;
    d[0] = f2bf(a.x + pz.x); d[1] = f2bf(a.y + pz.y);
    d[2] = f2bf(a.z + pz.z); d[3] = f2bf(a.w + pz.w);
  }
  __syncthreads();

  // ---- phase 2: Q/K/V projections (192 16x16 tiles, K=256 -> 8 wmma each) ----
  for (int tt = wv; tt < 192; tt += 8) {
    int mat = tt >> 6, rem = tt & 63;
    int mt = rem >> 4, nt = rem & 15;
    const bf16* W = (mat == 0) ? wq : (mat == 1) ? wk : wvv;
    v8f c = {};
#pragma unroll
    for (int kc = 0; kc < 8; ++kc) {
      v16bf a = ld_frag(s_tin, TS, mt * 16, kc * 32);
      v16bf b = ld_frag(W, 256, nt * 16, kc * 32);
      c = wmma_bf16(a, b, c);
    }
    if (mat == 0)      st_c_bf16(s_q, TS, mt * 16, nt * 16, c);
    else if (mat == 1) st_c_bf16(s_k, TS, mt * 16, nt * 16, c);
    else {                                           // store V transposed per head: vT[h][dh][t]
      int f = nt * 16 + (lane & 15);
      int head = f >> 5, dh = f & 31;
      int r0 = mt * 16 + ((lane >> 4) << 3);
      bf16* vt = s_vt + head * (32 * VTS) + dh * VTS;
#pragma unroll
      for (int i = 0; i < 8; ++i) vt[r0 + i] = f2bf(c[i]);
    }
  }
  __syncthreads();

  // ---- phase 3: scores = q·kT / 8, masked softmax in registers, attn -> bf16 LDS ----
  {
    int h = wv;                                      // wave per head
    bf16* attn_h = s_attn + h * (64 * AS);
    for (int mt = 0; mt < 4; ++mt) {
      v16bf aq = ld_frag(s_q, TS, mt * 16, h * 32);  // K = DH = 32: one wmma per tile
      v8f c[4];
#pragma unroll
      for (int st = 0; st < 4; ++st) {
        v16bf bk = ld_frag(s_k, TS, st * 16, h * 32);
        v8f z = {};
        c[st] = wmma_bf16(aq, bk, z);
      }
      int colb  = lane & 15;
      int rbase = mt * 16 + ((lane >> 4) << 3);
      float attnv[4][8];
#pragma unroll
      for (int i = 0; i < 8; ++i) {
        int t = rbase + i;
        int segT = s_seg[t];
        float vals[4], vmax = -3.0e38f;
#pragma unroll
        for (int st = 0; st < 4; ++st) {
          int s = st * 16 + colb;
          bool ok = (s <= t) && (s_seg[s] == segT);
          vals[st] = ok ? c[st][i] * 0.125f : -3.0e38f;   // 1/sqrt(64)
          vmax = fmaxf(vmax, vals[st]);
        }
#pragma unroll
        for (int off = 1; off < 16; off <<= 1) vmax = fmaxf(vmax, __shfl_xor(vmax, off, 32));
        float sum = 0.f;
#pragma unroll
        for (int st = 0; st < 4; ++st) { float e = __expf(vals[st] - vmax); attnv[st][i] = e; sum += e; }
#pragma unroll
        for (int off = 1; off < 16; off <<= 1) sum += __shfl_xor(sum, off, 32);
        float inv = 1.0f / sum;
#pragma unroll
        for (int st = 0; st < 4; ++st) attnv[st][i] *= inv;
      }
#pragma unroll
      for (int st = 0; st < 4; ++st) {
        v8f av;
#pragma unroll
        for (int i = 0; i < 8; ++i) av[i] = attnv[st][i];
        st_c_bf16(attn_h, AS, mt * 16, st * 16, av);
      }
    }
  }
  // (same-wave LDS RAW on s_attn; DS ops are in-order per wave, s_vt synced above)

  // ---- phase 4: out_h = attn·vT (K = T = 64 -> 2 wmma per tile) ----
  {
    int h = wv;
    const bf16* attn_h = s_attn + h * (64 * AS);
    const bf16* vt_h   = s_vt   + h * (32 * VTS);
    for (int mt = 0; mt < 4; ++mt)
      for (int nt = 0; nt < 2; ++nt) {
        v8f c = {};
#pragma unroll
        for (int kc = 0; kc < 2; ++kc) {
          v16bf a = ld_frag(attn_h, AS, mt * 16, kc * 32);
          v16bf b = ld_frag(vt_h, VTS, nt * 16, kc * 32);
          c = wmma_bf16(a, b, c);
        }
        st_c_bf16(s_out, TS, mt * 16, h * 32 + nt * 16, c);   // concat heads
      }
  }
  __syncthreads();

  // ---- phase 5: FF1  y1 = relu(out·WcT + bc) + out ----
  for (int tt = wv; tt < 64; tt += 8) {
    int mt = tt >> 4, nt = tt & 15;
    v8f c = {};
#pragma unroll
    for (int kc = 0; kc < 8; ++kc) {
      v16bf a = ld_frag(s_out, TS, mt * 16, kc * 32);
      v16bf b = ld_frag(wc, 256, nt * 16, kc * 32);
      c = wmma_bf16(a, b, c);
    }
    int col = nt * 16 + (lane & 15);
    int r0  = mt * 16 + ((lane >> 4) << 3);
    float bcv = bc[col];
#pragma unroll
    for (int i = 0; i < 8; ++i) {
      int t = r0 + i;
      float y = fmaxf(c[i] + bcv, 0.f) + bf2f(s_out[t * TS + col]);
      s_y1[t * TS + col] = f2bf(y);
    }
  }
  __syncthreads();

  // ---- phase 6: FF2  y2 = relu(y1·WcT + bc) + y1  (f32 result for LN) ----
  for (int tt = wv; tt < 64; tt += 8) {
    int mt = tt >> 4, nt = tt & 15;
    v8f c = {};
#pragma unroll
    for (int kc = 0; kc < 8; ++kc) {
      v16bf a = ld_frag(s_y1, TS, mt * 16, kc * 32);
      v16bf b = ld_frag(wc, 256, nt * 16, kc * 32);
      c = wmma_bf16(a, b, c);
    }
    int col = nt * 16 + (lane & 15);
    int r0  = mt * 16 + ((lane >> 4) << 3);
    float bcv = bc[col];
#pragma unroll
    for (int i = 0; i < 8; ++i) {
      int t = r0 + i;
      float y = fmaxf(c[i] + bcv, 0.f) + bf2f(s_y1[t * TS + col]);
      s_y2[t * Y2S + col] = y;
    }
  }
  __syncthreads();

  // ---- phase 7: LayerNorm per row, write fp32 output ----
  {
    int r = tid >> 2, part = tid & 3;                // 4 threads per row, 64 elems each
    const float* row = s_y2 + r * Y2S;
    float sum = 0.f, sq = 0.f;
#pragma unroll 8
    for (int j = 0; j < 64; ++j) { float x = row[part * 64 + j]; sum += x; sq += x * x; }
    sum += __shfl_xor(sum, 1, 32); sum += __shfl_xor(sum, 2, 32);
    sq  += __shfl_xor(sq,  1, 32); sq  += __shfl_xor(sq,  2, 32);
    float m    = sum * (1.0f / 256.0f);
    float var  = sq * (1.0f / 256.0f) - m * m;
    float rstd = rsqrtf(var + 1e-5f);
    float* orow = out + ((size_t)n * 64 + r) * 256;
#pragma unroll
    for (int j = 0; j < 64; j += 4) {
      int ci = part * 64 + j;
      float4 o;
      o.x = (row[ci + 0] - m) * rstd * g[ci + 0] + bln[ci + 0];
      o.y = (row[ci + 1] - m) * rstd * g[ci + 1] + bln[ci + 1];
      o.z = (row[ci + 2] - m) * rstd * g[ci + 2] + bln[ci + 2];
      o.w = (row[ci + 3] - m) * rstd * g[ci + 3] + bln[ci + 3];
      *(float4*)(orow + ci) = o;
    }
  }
}

// ---------------- launch ----------------
extern "C" void kernel_launch(void* const* d_in, const int* in_sizes, int n_in,
                              void* d_out, int out_size, void* d_ws, size_t ws_size,
                              hipStream_t stream) {
  (void)in_sizes; (void)n_in; (void)out_size; (void)ws_size;
  const float*     inputs = (const float*)d_in[0];
  const long long* seg    = (const long long*)d_in[1];
  const float*     pos    = (const float*)d_in[2];
  const float*     Wq     = (const float*)d_in[3];
  const float*     Wk     = (const float*)d_in[4];
  const float*     Wv     = (const float*)d_in[5];
  const float*     Wc     = (const float*)d_in[6];
  const float*     bc     = (const float*)d_in[7];
  const float*     g      = (const float*)d_in[8];
  const float*     bl     = (const float*)d_in[9];
  bf16* wbf = (bf16*)d_ws;                         // 512 KB bf16 weights, stays in L2

  wconv_kernel<<<1024, 256, 0, stream>>>(Wq, Wk, Wv, Wc, wbf);
  attn_fused_kernel<<<4096, 256, SMEM_BYTES, stream>>>(inputs, seg, pos, wbf,
                                                       bc, g, bl, (float*)d_out);
}
